// SmyrfAttention_47304769798230
// MI455X (gfx1250) — matrix operand order repaired and checked by reading, not compiled, not run on hardware
//
#include <hip/hip_runtime.h>
#include <hip/hip_bf16.h>
#include <math.h>

// Problem constants (from reference setup_inputs)
#define NH   8      // n_hashes
#define BB   2      // batch
#define TT   8192   // T == S
#define HH   8      // heads
#define EE   64     // E == D
#define DD   64
#define BHC  16     // B*H
#define BUCKET 128
#define NB   64     // TT / BUCKET
#define SORTROWS 256 // 2 * NH * BHC  (q rows then k rows)

typedef __attribute__((ext_vector_type(16))) _Float16 v16h_t;
typedef __attribute__((ext_vector_type(8)))  _Float16 v8h_t;
typedef __attribute__((ext_vector_type(8)))  float    v8f_t;

// ---- WMMA fragment loaders from LDS (row-major tiles) ------------------
// A-matrix 16x32 f16 (ISA 7.12.2): lane l -> row M=l&15; hi=(l>>4);
//   halves j=0..7  : K = kk + hi*8 + j
//   halves j=8..15 : K = kk + 16 + hi*8 + (j-8)
__device__ __forceinline__ v16h_t ld_fragA(const _Float16* rowp, int kk, int hi8) {
  v8h_t a = *(const v8h_t*)(rowp + kk + hi8);
  v8h_t b = *(const v8h_t*)(rowp + kk + 16 + hi8);
  return __builtin_shufflevector(a, b, 0,1,2,3,4,5,6,7,8,9,10,11,12,13,14,15);
}
// B-matrix 32x16 f16 (same K order as A):
//   lane l -> col N=l&15; hi=(l>>4); halves j=0..7 : K = kk + hi*8 + j; j=8..15 : K = kk + 16 + hi*8 + (j-8)
__device__ __forceinline__ v16h_t ld_fragB(const _Float16* rowp, int kk, int hi16) {
  v8h_t a = *(const v8h_t*)(rowp + kk + hi16);
  v8h_t b = *(const v8h_t*)(rowp + kk + 16 + hi16);
  return __builtin_shufflevector(a, b, 0,1,2,3,4,5,6,7,8,9,10,11,12,13,14,15);
}

// ---------------------------------------------------------------------
// Kernel 0: init accumulators + norm maxima
__global__ void smyrf_init_kernel(float* acc_m, float* acc_s, float* acc_o,
                                  int* maxq2i, int* maxk2i) {
  size_t i = (size_t)blockIdx.x * blockDim.x + threadIdx.x;
  if (i < (size_t)BHC * TT) { acc_m[i] = -1e30f; acc_s[i] = 0.f; }
  if (i < (size_t)BHC * TT * DD) acc_o[i] = 0.f;
  if (i < BHC) { maxq2i[i] = 0; maxk2i[i] = 0; }
}

// ---------------------------------------------------------------------
// Kernel 1: per-row squared norms + per-bh max (positive-float atomicMax on bits)
__global__ __launch_bounds__(256)
void smyrf_norms_kernel(const float* __restrict__ query, const float* __restrict__ key,
                        float* qn2, float* kn2, int* maxq2i, int* maxk2i) {
  int blk = blockIdx.x;                 // 2 * BHC * 32 blocks
  int isK = blk >= BHC * 32;
  if (isK) blk -= BHC * 32;
  int bh = blk / 32;
  int t  = (blk % 32) * 256 + threadIdx.x;
  int b = bh / HH, hh = bh % HH;
  const float* rowp = (isK ? key : query) + (((size_t)b * TT + t) * HH + hh) * EE;
  float s = 0.f;
  #pragma unroll 8
  for (int e = 0; e < EE; ++e) { float v = rowp[e]; s += v * v; }
  (isK ? kn2 : qn2)[(size_t)bh * TT + t] = s;
  __shared__ float red[256];
  red[threadIdx.x] = s;
  __syncthreads();
  for (int o = 128; o > 0; o >>= 1) {
    if (threadIdx.x < o) red[threadIdx.x] = fmaxf(red[threadIdx.x], red[threadIdx.x + o]);
    __syncthreads();
  }
  if (threadIdx.x == 0)
    atomicMax(isK ? &maxk2i[bh] : &maxq2i[bh], __float_as_int(red[0]));
}

// ---------------------------------------------------------------------
// Kernel 2: E2LSH hash values with XBOX+ extension column
//   qh row (h*BHC+bh), kh row (128 + h*BHC+bh), each length TT
__global__ __launch_bounds__(256)
void smyrf_hash_kernel(const float* __restrict__ query, const float* __restrict__ key,
                       const float* __restrict__ alpha, const float* __restrict__ beta,
                       const float* __restrict__ qn2, const float* __restrict__ kn2,
                       const int* maxq2i, const int* maxk2i, float* hashBuf) {
  __shared__ float alphaS[(EE + 2) * NH];
  __shared__ float betaS[NH];
  for (int i = threadIdx.x; i < (EE + 2) * NH; i += blockDim.x) alphaS[i] = alpha[i];
  if (threadIdx.x < NH) betaS[threadIdx.x] = beta[threadIdx.x];
  __syncthreads();

  size_t gid = (size_t)blockIdx.x * blockDim.x + threadIdx.x;   // 0 .. 2*BHC*TT
  int isK = gid >= (size_t)BHC * TT;
  size_t r = isK ? gid - (size_t)BHC * TT : gid;
  int bh = (int)(r / TT), t = (int)(r % TT);
  int b = bh / HH, hh = bh % HH;
  const float* rowp = (isK ? key : query) + (((size_t)b * TT + t) * HH + hh) * EE;

  float M2 = __int_as_float(maxq2i[bh]) + __int_as_float(maxk2i[bh]);
  float n2 = (isK ? kn2 : qn2)[(size_t)bh * TT + t];
  float ext = sqrtf(fmaxf(M2 - n2, 0.f));

  float acc[NH];
  #pragma unroll
  for (int h = 0; h < NH; ++h) acc[h] = betaS[h];
  for (int e = 0; e < EE; ++e) {
    float v = rowp[e];
    #pragma unroll
    for (int h = 0; h < NH; ++h) acc[h] += v * alphaS[e * NH + h];
  }
  int extcol = isK ? (EE + 1) : EE;
  #pragma unroll
  for (int h = 0; h < NH; ++h) acc[h] += ext * alphaS[extcol * NH + h];

  size_t rowBase = (size_t)(isK ? (NH * BHC) : 0);
  #pragma unroll
  for (int h = 0; h < NH; ++h)
    hashBuf[(rowBase + (size_t)h * BHC + bh) * TT + t] = acc[h];
}

// ---------------------------------------------------------------------
// Kernel 3: in-LDS bitonic argsort of 8192 keys per row (ascending)
__global__ __launch_bounds__(1024)
void smyrf_sort_kernel(const float* __restrict__ hashBuf, int* __restrict__ posBuf) {
  extern __shared__ __align__(16) char smem[];
  float* keys = (float*)smem;              // 8192 floats
  int*   idx  = (int*)(smem + TT * 4);     // 8192 ints
  int row = blockIdx.x;
  const float* src = hashBuf + (size_t)row * TT;
  for (int i = threadIdx.x; i < TT; i += blockDim.x) { keys[i] = src[i]; idx[i] = i; }
  __syncthreads();
  for (int k = 2; k <= TT; k <<= 1) {
    for (int j = k >> 1; j > 0; j >>= 1) {
      for (int i = threadIdx.x; i < TT; i += blockDim.x) {
        int ixj = i ^ j;
        if (ixj > i) {
          bool up = ((i & k) == 0);
          float ki = keys[i], kj = keys[ixj];
          if ((ki > kj) == up) {
            keys[i] = kj; keys[ixj] = ki;
            int tmp = idx[i]; idx[i] = idx[ixj]; idx[ixj] = tmp;
          }
        }
      }
      __syncthreads();
    }
  }
  int* dst = posBuf + (size_t)row * TT;
  for (int i = threadIdx.x; i < TT; i += blockDim.x) dst[i] = idx[i];
}

// ---------------------------------------------------------------------
// Kernel 4 (launched once per hash): bucketed 128x128 WMMA attention with
// streaming logsumexp merge into (acc_m, acc_s, acc_o).
// Dynamic LDS layout (bytes):
//   Qs  f16 [128][64]   @ 0       (16384)
//   Ks  f16 [128][64]   @ 16384   (16384)
//   Vt  f16 [64][128]   @ 32768   (16384)   (transposed V: Vt[d][krow])
//   Ps  f16 [128][128]  @ 49152   (32768)
//   Sld f32 [128][128]  @ 81920   (65536)   (aliased later by Old f32 [128][64])
#define ATTN_LDS_BYTES (16384 + 16384 + 16384 + 32768 + 65536)

__global__ __launch_bounds__(256)
void smyrf_attn_kernel(const float* __restrict__ query, const float* __restrict__ key,
                       const float* __restrict__ value, const int* __restrict__ posBuf,
                       float* acc_m, float* acc_s, float* acc_o, int hashIdx) {
  extern __shared__ __align__(16) char smem[];
  _Float16* Qs = (_Float16*)smem;
  _Float16* Ks = (_Float16*)(smem + 16384);
  _Float16* Vt = (_Float16*)(smem + 32768);
  _Float16* Ps = (_Float16*)(smem + 49152);
  float*    Sld = (float*)(smem + 81920);
  float*    Old = Sld;                      // reuse score buffer for O strip

  __shared__ int   tArr[128];
  __shared__ float lseArr[128];
  __shared__ float invS[128];
  __shared__ float rowAlpha[128];
  __shared__ float rowW[128];

  const int tid = threadIdx.x;
  const int bh = blockIdx.x >> 6;           // 64 buckets per (bh)
  const int bucket = blockIdx.x & 63;
  const int b = bh >> 3, hh = bh & 7;

  const int* qpos = posBuf + ((size_t)(hashIdx * BHC + bh)) * TT + bucket * BUCKET;
  const int* kpos = posBuf + ((size_t)(NH * BHC + hashIdx * BHC + bh)) * TT + bucket * BUCKET;

  // ---- gather bucket tiles into LDS (f32 -> f16) ----
  for (int idx = tid; idx < BUCKET * EE; idx += 256) {
    int r = idx >> 6, e = idx & 63;
    int tq = qpos[r];
    Qs[idx] = (_Float16)query[(((size_t)b * TT + tq) * HH + hh) * EE + e];
    if (e == 0) tArr[r] = tq;
    int tk = kpos[r];
    const size_t kbase = (((size_t)b * TT + tk) * HH + hh) * EE;
    Ks[idx] = (_Float16)key[kbase + e];
    Vt[e * BUCKET + r] = (_Float16)value[kbase + e];
  }
  __syncthreads();

  const int wave = tid >> 5, lane = tid & 31;
  const int m0 = wave << 4;
  const int l15 = lane & 15;
  const int hi8  = (lane >> 4) << 3;
  const int hi16 = (lane >> 4) << 4;

  // ---- S = Q * K^T : each wave owns rows [m0, m0+16), all 128 cols ----
  v8f_t accS[8] = {};
  #pragma unroll
  for (int kk = 0; kk < EE; kk += 32) {
    v16h_t afr = ld_fragA(Qs + (m0 + l15) * EE, kk, hi8);
    #pragma unroll
    for (int n = 0; n < 8; ++n) {
      v16h_t bfr = ld_fragB(Ks + (n * 16 + l15) * EE, kk, hi8);
      accS[n] = __builtin_amdgcn_wmma_f32_16x16x32_f16(
          false, afr, false, bfr, (short)0, accS[n], false, false);
    }
  }
  #pragma unroll
  for (int n = 0; n < 8; ++n)
    #pragma unroll
    for (int v = 0; v < 8; ++v)
      Sld[(m0 + v + hi8) * 128 + n * 16 + l15] = accS[n][v];
  __syncthreads();

  // ---- row softmax (unnormalized P = exp(x - m); lse = m + log s) ----
  const float temp = 0.125f; // 1/sqrt(64)
  if (tid < 128) {
    int r = tid;
    int rot = r & 127;
    float m = -1e30f;
    for (int c = 0; c < 128; ++c) {
      int cc = (c + rot) & 127;
      m = fmaxf(m, Sld[r * 128 + cc] * temp);
    }
    float s = 0.f;
    for (int c = 0; c < 128; ++c) {
      int cc = (c + rot) & 127;
      float p = __expf(Sld[r * 128 + cc] * temp - m);
      s += p;
      Ps[r * 128 + cc] = (_Float16)p;
    }
    lseArr[r] = m + __logf(s);
    invS[r] = 1.f / s;
  }
  __syncthreads();

  // ---- O = P * V (unnormalized) ----
  v8f_t accO[4] = {};
  #pragma unroll
  for (int kk = 0; kk < BUCKET; kk += 32) {
    v16h_t afr = ld_fragA(Ps + (m0 + l15) * 128, kk, hi8);
    #pragma unroll
    for (int n = 0; n < 4; ++n) {
      v16h_t bfr = ld_fragB(Vt + (n * 16 + l15) * BUCKET, kk, hi8);
      accO[n] = __builtin_amdgcn_wmma_f32_16x16x32_f16(
          false, afr, false, bfr, (short)0, accO[n], false, false);
    }
  }
  #pragma unroll
  for (int n = 0; n < 4; ++n)
    #pragma unroll
    for (int v = 0; v < 8; ++v)
      Old[(m0 + v + hi8) * DD + n * 16 + l15] = accO[n][v];
  __syncthreads();

  // ---- streaming logsumexp merge (per-row scalars, then scatter o) ----
  if (tid < 128) {
    int r = tid;
    size_t g = (size_t)bh * TT + tArr[r];
    float lse = lseArr[r];
    float mo = acc_m[g], so = acc_s[g];
    float mn = fmaxf(mo, lse);
    float ao = __expf(mo - mn);
    float w  = __expf(lse - mn);
    acc_m[g] = mn;
    acc_s[g] = so * ao + w;
    rowAlpha[r] = ao;
    rowW[r] = w * invS[r];   // normalize O inside merge weight
  }
  __syncthreads();
  for (int idx = tid; idx < BUCKET * DD; idx += 256) {
    int r = idx >> 6, d = idx & 63;
    size_t g = ((size_t)bh * TT + tArr[r]) * DD + d;
    acc_o[g] = acc_o[g] * rowAlpha[r] + Old[idx] * rowW[r];
  }
}

// ---------------------------------------------------------------------
// Kernel 5: out[b][t][h][d] = acc_o[bh][t][d] / acc_s[bh][t]
__global__ void smyrf_final_kernel(const float* __restrict__ acc_o,
                                   const float* __restrict__ acc_s,
                                   float* __restrict__ out) {
  size_t i = (size_t)blockIdx.x * blockDim.x + threadIdx.x; // B*T*H*D
  int d = (int)(i & 63);
  size_t j = i >> 6;
  int hh = (int)(j & 7);
  size_t j2 = j >> 3;
  int t = (int)(j2 & (TT - 1));
  int b = (int)(j2 >> 13);
  int bh = b * HH + hh;
  float s = acc_s[(size_t)bh * TT + t];
  out[i] = acc_o[((size_t)bh * TT + t) * DD + d] / s;
}

// ---------------------------------------------------------------------
extern "C" void kernel_launch(void* const* d_in, const int* in_sizes, int n_in,
                              void* d_out, int out_size, void* d_ws, size_t ws_size,
                              hipStream_t stream) {
  const float* query = (const float*)d_in[0];  // (B,T,H,E)
  const float* key   = (const float*)d_in[1];  // (B,S,H,E)
  const float* value = (const float*)d_in[2];  // (B,S,H,D)
  const float* alpha = (const float*)d_in[3];  // (E+2, NH)
  const float* beta  = (const float*)d_in[4];  // (1, NH)
  float* out = (float*)d_out;

  // workspace carve-up (all 4-byte elements)
  float* ws = (float*)d_ws;
  float* qn2 = ws;                                   // BHC*TT
  float* kn2 = qn2 + (size_t)BHC * TT;               // BHC*TT
  int*   maxq2i = (int*)(kn2 + (size_t)BHC * TT);    // BHC
  int*   maxk2i = maxq2i + BHC;                      // BHC
  float* hashBuf = (float*)(maxk2i + BHC);           // SORTROWS*TT
  int*   posBuf = (int*)(hashBuf + (size_t)SORTROWS * TT); // SORTROWS*TT
  float* acc_m = (float*)(posBuf + (size_t)SORTROWS * TT); // BHC*TT
  float* acc_s = acc_m + (size_t)BHC * TT;           // BHC*TT
  float* acc_o = acc_s + (size_t)BHC * TT;           // BHC*TT*DD

  // 0) init accumulators / maxima
  {
    size_t n = (size_t)BHC * TT * DD;
    smyrf_init_kernel<<<dim3((unsigned)((n + 255) / 256)), dim3(256), 0, stream>>>(
        acc_m, acc_s, acc_o, maxq2i, maxk2i);
  }
  // 1) norms + maxima
  smyrf_norms_kernel<<<dim3(2 * BHC * 32), dim3(256), 0, stream>>>(
      query, key, qn2, kn2, maxq2i, maxk2i);
  // 2) hashes
  smyrf_hash_kernel<<<dim3((2 * BHC * TT) / 256), dim3(256), 0, stream>>>(
      query, key, alpha, beta, qn2, kn2, maxq2i, maxk2i, hashBuf);
  // 3) argsort each of 256 rows (64KB dynamic LDS)
  smyrf_sort_kernel<<<dim3(SORTROWS), dim3(1024), TT * 8, stream>>>(hashBuf, posBuf);
  // 4) bucketed WMMA attention, one launch per hash (stream-ordered merge)
  for (int h = 0; h < NH; ++h) {
    smyrf_attn_kernel<<<dim3(BHC * NB), dim3(256), ATTN_LDS_BYTES, stream>>>(
        query, key, value, posBuf, acc_m, acc_s, acc_o, h);
  }
  // 5) normalize + layout to (B,T,H,D)
  {
    size_t n = (size_t)BB * TT * HH * DD;
    smyrf_final_kernel<<<dim3((unsigned)((n + 255) / 256)), dim3(256), 0, stream>>>(
        acc_o, acc_s, out);
  }
  (void)in_sizes; (void)n_in; (void)out_size; (void)ws_size;
}
